// MaskedAsymmetricHistogramLoss_32083405701207
// MI455X (gfx1250) — compile-verified, matrix operand order checked by reading, not run on hardware
//
#include <hip/hip_runtime.h>
#include <hip/hip_bf16.h>
#include <hip/hip_fp16.h>

typedef __attribute__((ext_vector_type(16))) _Float16 v16h;
typedef __attribute__((ext_vector_type(8)))  float    v8f;

#define BATCH 16
#define CH    3
#define H     256
#define W     256
#define HD    128
#define WD    128
#define NPIX  (HD*WD)      // 16384
#define NBINS 64
#define DIL   8
#define CHUNK 2048         // pixels staged in LDS per stage

__device__ __forceinline__ int refl(int i, int n) {
  // jnp.pad mode='reflect', pad=1: -1 -> 1, n -> n-2
  return i < 0 ? -i : (i >= n ? 2*n - 2 - i : i);
}

// ---------------------------------------------------------------------------
// Kernel 1: Sobel edge magnitude (per channel, reflect pad), max over C,
// threshold -> binary core mask (B,256,256)
// ---------------------------------------------------------------------------
__global__ void k_edge_core(const float* __restrict__ tgt, float* __restrict__ core) {
  int tid = blockIdx.x * 256 + threadIdx.x;
  if (tid >= BATCH * H * W) return;
  int x = tid % W;
  int y = (tid / W) % H;
  int b = tid / (H * W);
  float mmax = 0.0f;
  for (int c = 0; c < CH; ++c) {
    const float* p = tgt + ((size_t)(b * CH + c)) * H * W;
    float v[3][3];
#pragma unroll
    for (int dy = -1; dy <= 1; ++dy)
#pragma unroll
      for (int dx = -1; dx <= 1; ++dx)
        v[dy + 1][dx + 1] = p[refl(y + dy, H) * W + refl(x + dx, W)];
    // cross-correlation with SOBEL / SOBEL^T (XLA conv does not flip kernels)
    float gx = (v[0][2] - v[0][0]) + 2.0f * (v[1][2] - v[1][0]) + (v[2][2] - v[2][0]);
    float gy = (v[2][0] - v[0][0]) + 2.0f * (v[2][1] - v[0][1]) + (v[2][2] - v[0][2]);
    float mag = sqrtf(gx * gx + gy * gy + 1e-8f);
    mmax = fmaxf(mmax, mag);
  }
  core[tid] = (mmax > 0.05f) ? 1.0f : 0.0f;
}

// ---------------------------------------------------------------------------
// Kernel 2: horizontal 17-wide max, evaluated only at even x (x = 2j)
// core(B,256,256) -> rowmax(B,256,128)
// ---------------------------------------------------------------------------
__global__ void k_dilate_h(const float* __restrict__ core, float* __restrict__ rowmax) {
  int tid = blockIdx.x * 256 + threadIdx.x;
  if (tid >= BATCH * H * WD) return;
  int j = tid % WD;
  int y = (tid / WD) % H;
  int b = tid / (H * WD);
  int x0 = max(0, 2 * j - DIL);
  int x1 = min(W - 1, 2 * j + DIL);
  const float* row = core + ((size_t)(b * H + y)) * W;
  float m = 0.0f;
  for (int x = x0; x <= x1; ++x) m = fmaxf(m, row[x]);
  rowmax[tid] = m;
}

// ---------------------------------------------------------------------------
// Kernel 3: vertical 17-wide max at even y (y = 2i)
// rowmax(B,256,128) -> mask_down(B,128,128)
// ---------------------------------------------------------------------------
__global__ void k_dilate_v(const float* __restrict__ rowmax, float* __restrict__ maskd) {
  int tid = blockIdx.x * 256 + threadIdx.x;
  if (tid >= BATCH * NPIX) return;
  int j = tid % WD;
  int i = (tid / WD) % HD;
  int b = tid / NPIX;
  int y0 = max(0, 2 * i - DIL);
  int y1 = min(H - 1, 2 * i + DIL);
  float m = 0.0f;
  for (int y = y0; y <= y1; ++y) m = fmaxf(m, rowmax[((size_t)(b * H + y)) * WD + j]);
  maskd[tid] = m;
}

// ---------------------------------------------------------------------------
// Kernel 4: pred 2x2 avg-pool; target 3x3 Gaussian blur (reflect) + 2x2 avg.
// Both clipped to [0,1] (reference clips before histogram).
// ---------------------------------------------------------------------------
__global__ void k_downsample(const float* __restrict__ pred, const float* __restrict__ tgt,
                             float* __restrict__ pdown, float* __restrict__ tdown) {
  int tid = blockIdx.x * 256 + threadIdx.x;
  if (tid >= BATCH * CH * NPIX) return;
  int j  = tid % WD;
  int i  = (tid / WD) % HD;
  int bc = tid / NPIX;
  const float* pp = pred + (size_t)bc * H * W;
  const float* tp = tgt  + (size_t)bc * H * W;
  int y0 = 2 * i, x0 = 2 * j;

  float ps = pp[y0 * W + x0] + pp[y0 * W + x0 + 1]
           + pp[(y0 + 1) * W + x0] + pp[(y0 + 1) * W + x0 + 1];
  pdown[tid] = fminf(fmaxf(0.25f * ps, 0.0f), 1.0f);

  // normalized separable Gaussian: [a,1,a]/(1+2a), a = exp(-0.5)
  const float w0 = 0.27406862f, w1 = 0.45186276f;
  const float wv[3] = {w0, w1, w0};
  float ts = 0.0f;
#pragma unroll
  for (int sy = 0; sy < 2; ++sy)
#pragma unroll
    for (int sx = 0; sx < 2; ++sx) {
      float acc = 0.0f;
#pragma unroll
      for (int dy = -1; dy <= 1; ++dy)
#pragma unroll
        for (int dx = -1; dx <= 1; ++dx)
          acc += wv[dy + 1] * wv[dx + 1] *
                 tp[refl(y0 + sy + dy, H) * W + refl(x0 + sx + dx, W)];
      ts += acc;
    }
  tdown[tid] = fminf(fmaxf(0.25f * ts, 0.0f), 1.0f);
}

// ---------------------------------------------------------------------------
// Kernel 5: masked soft histogram via WMMA reduction.
// One block per (b,c); 8 waves: wave = {pred|gt} x {bin group of 16}.
// Per 32-pixel step: B = 32x16 f16 exp-weights (pixels x bins),
// A = 16x32 f16 with mask in row 0, zeros elsewhere; D=A*B+C accumulates
// the 16-bin histogram in row 0 of the f32 accumulator (VGPR0, lanes 0..15).
// CDNA5 16-bit A/B element mapping: lane L -> {M|N}=L%16, g=L>>4,
//   slot j -> K = 16*(j>>3) + 8*g + (j&7).
// ---------------------------------------------------------------------------
__global__ void __launch_bounds__(256) k_hist_wmma(const float* __restrict__ pdown,
                                                   const float* __restrict__ tdown,
                                                   const float* __restrict__ maskd,
                                                   float* __restrict__ hp,
                                                   float* __restrict__ hg) {
  __shared__ float xs_p[CHUNK];
  __shared__ float xs_t[CHUNK];
  __shared__ float ms[CHUNK];

  int bc   = blockIdx.x;      // 0..47
  int b    = bc / CH;
  int tid  = threadIdx.x;
  int lane = tid & 31;
  int wave = tid >> 5;
  int hist_sel = wave >> 2;   // 0 = pred, 1 = gt
  int bg       = wave & 3;    // bin group (16 bins each)
  int n        = lane & 15;   // output column = bin within group; also A row M
  int g        = lane >> 4;   // K-half selector
  float binv   = (float)(bg * 16 + n) * (1.0f / 63.0f);
  float rowsel = (n == 0) ? 1.0f : 0.0f;  // mask only feeds A row 0

  v8f acc = {};
  for (int stage = 0; stage < NPIX / CHUNK; ++stage) {
    int base = stage * CHUNK;
    for (int t = tid; t < CHUNK; t += 256) {
      xs_p[t] = pdown[(size_t)bc * NPIX + base + t];
      xs_t[t] = tdown[(size_t)bc * NPIX + base + t];
      ms[t]   = maskd[(size_t)b  * NPIX + base + t];
    }
    __syncthreads();
    const float* xs = hist_sel ? xs_t : xs_p;
    for (int pos = 0; pos < CHUNK; pos += 32) {
      v16h av, bv;
#pragma unroll
      for (int jj = 0; jj < 16; ++jj) {
        int K   = ((jj >> 3) << 4) + (g << 3) + (jj & 7);
        float x = xs[pos + K];
        float d = x - binv;
        float wgt = __expf(d * d * -1250.0f);   // 1/(2*sigma^2) = 1250
        bv[jj] = (_Float16)wgt;
        av[jj] = (_Float16)(rowsel * ms[pos + K]);
      }
      acc = __builtin_amdgcn_wmma_f32_16x16x32_f16(
          /*neg_a=*/false, av, /*neg_b=*/false, bv,
          /*c_mod=*/(short)0, acc, /*reuse_a=*/false, /*reuse_b=*/false);
    }
    __syncthreads();
  }
  // f32 C/D layout: VGPR0 holds M=0 for lanes 0..15 (N = lane).
  if (lane < 16) {
    float* out = hist_sel ? hg : hp;
    out[bc * NBINS + bg * 16 + lane] = acc[0];
  }
}

// ---------------------------------------------------------------------------
// Kernel 6: normalize by per-batch mask sums, asymmetric relu means -> scalar
// ---------------------------------------------------------------------------
__global__ void __launch_bounds__(256) k_finalize(const float* __restrict__ maskd,
                                                  const float* __restrict__ hp,
                                                  const float* __restrict__ hg,
                                                  float* __restrict__ out) {
  __shared__ float sdata[256];
  __shared__ float msum[BATCH];
  int tid = threadIdx.x;

  for (int b = 0; b < BATCH; ++b) {
    float local = 0.0f;
    for (int i = tid; i < NPIX; i += 256) local += maskd[(size_t)b * NPIX + i];
    sdata[tid] = local; __syncthreads();
    for (int s = 128; s > 0; s >>= 1) {
      if (tid < s) sdata[tid] += sdata[tid + s];
      __syncthreads();
    }
    if (tid == 0) msum[b] = sdata[0];
    __syncthreads();
  }

  float ex = 0.0f, mi = 0.0f;
  for (int idx = tid; idx < BATCH * CH * NBINS; idx += 256) {
    int bc = idx >> 6;
    int b  = bc / CH;
    float d = (hp[idx] - hg[idx]) / (msum[b] + 1e-8f);
    ex += fmaxf(d, 0.0f);
    mi += fmaxf(-d, 0.0f);
  }
  sdata[tid] = ex; __syncthreads();
  for (int s = 128; s > 0; s >>= 1) { if (tid < s) sdata[tid] += sdata[tid + s]; __syncthreads(); }
  float Ex = sdata[0]; __syncthreads();
  sdata[tid] = mi; __syncthreads();
  for (int s = 128; s > 0; s >>= 1) { if (tid < s) sdata[tid] += sdata[tid + s]; __syncthreads(); }
  if (tid == 0) {
    float inv = 1.0f / (float)(BATCH * CH * NBINS);
    out[0] = 5.0f * Ex * inv + sdata[0] * inv;   // W_EXTRA=5, W_MISSING=1
  }
}

extern "C" void kernel_launch(void* const* d_in, const int* in_sizes, int n_in,
                              void* d_out, int out_size, void* d_ws, size_t ws_size,
                              hipStream_t stream) {
  (void)in_sizes; (void)n_in; (void)out_size; (void)ws_size;
  const float* pred = (const float*)d_in[0];
  const float* tgt  = (const float*)d_in[1];
  float* out = (float*)d_out;

  float* ws     = (float*)d_ws;
  float* core   = ws;                            // B*H*W          = 1,048,576 f32
  float* rowmax = core   + BATCH * H * W;        // B*H*WD         =   524,288
  float* maskd  = rowmax + BATCH * H * WD;       // B*NPIX         =   262,144
  float* pdown  = maskd  + BATCH * NPIX;         // B*C*NPIX       =   786,432
  float* tdown  = pdown  + BATCH * CH * NPIX;    // B*C*NPIX       =   786,432
  float* hp     = tdown  + BATCH * CH * NPIX;    // B*C*NBINS      =     3,072
  float* hg     = hp     + BATCH * CH * NBINS;   // B*C*NBINS      =     3,072

  k_edge_core <<<(BATCH * H * W       + 255) / 256, 256, 0, stream>>>(tgt, core);
  k_dilate_h  <<<(BATCH * H * WD      + 255) / 256, 256, 0, stream>>>(core, rowmax);
  k_dilate_v  <<<(BATCH * NPIX        + 255) / 256, 256, 0, stream>>>(rowmax, maskd);
  k_downsample<<<(BATCH * CH * NPIX   + 255) / 256, 256, 0, stream>>>(pred, tgt, pdown, tdown);
  k_hist_wmma <<<BATCH * CH, 256, 0, stream>>>(pdown, tdown, maskd, hp, hg);
  k_finalize  <<<1, 256, 0, stream>>>(maskd, hp, hg, out);
}
